// Attention_Layer_66417374266003
// MI455X (gfx1250) — compile-verified, matrix operand order checked by reading, not run on hardware
//
#include <hip/hip_runtime.h>
#include <hip/hip_bf16.h>

// ---------------------------------------------------------------------------
// Attention layer for MI455X (gfx1250, wave32, WMMA).
//   B=4, S=2048, D=1024
//   y = softmax((x@Wq+bq)(x@Wk+bk)^T / sqrt(D)) @ (x@Wv+bv)
//
// All matmuls: v_wmma_f32_16x16x32_f16 on a shared double-buffered GEMM core.
// LDS staging uses GLOBAL_LOAD_ASYNC_TO_LDS_B128 (ASYNCcnt) when the builtin
// exists; otherwise falls back to synchronous 128-bit copies.
// ---------------------------------------------------------------------------

typedef __attribute__((ext_vector_type(16))) _Float16 v16h;
typedef __attribute__((ext_vector_type(8)))  float    v8f;
typedef __attribute__((ext_vector_type(4)))  int      v4i;

#define BATCH 4
#define SEQ   2048
#define DIM   1024

#define TILE_HALVES (128 * 32)   // one [128 rows][32 halves] LDS tile

#if defined(__HIP_DEVICE_COMPILE__) && \
    __has_builtin(__builtin_amdgcn_global_load_async_to_lds_b128)
#define USE_ASYNC_LDS 1
#else
#define USE_ASYNC_LDS 0
#endif

typedef __attribute__((address_space(1))) v4i gas_v4i;   // global int4*
typedef __attribute__((address_space(3))) v4i las_v4i;   // LDS int4*

__device__ __forceinline__ void async_wait_all() {
#if USE_ASYNC_LDS
#if __has_builtin(__builtin_amdgcn_s_wait_asynccnt)
    __builtin_amdgcn_s_wait_asynccnt(0);
#else
    asm volatile("s_wait_asynccnt 0" ::: "memory");
#endif
#endif
}

// ---------------------------------------------------------------------------
// Stage one [128][32] f16 tile into LDS. 512 16-byte chunks / 256 threads.
// Async path: per-lane LDS-DMA, tracked by ASYNCcnt (no VGPR data movement).
// ---------------------------------------------------------------------------
__device__ __forceinline__ void stage_tile(_Float16* lds, const _Float16* g,
                                           int ldg, int t) {
#pragma unroll
    for (int c2 = 0; c2 < 2; ++c2) {
        const int c = t + c2 * 256;
        const int row = c >> 2, col = (c & 3) << 3;   // col in halves, 16B chunks
        const _Float16* gp = g + (size_t)row * ldg + col;
        _Float16* lp = lds + row * 32 + col;
#if USE_ASYNC_LDS
        __builtin_amdgcn_global_load_async_to_lds_b128(
            (gas_v4i*)gp, (las_v4i*)lp, 0, 0);
#else
        *reinterpret_cast<uint4*>(lp) = *reinterpret_cast<const uint4*>(gp);
#endif
    }
}

// ---------------------------------------------------------------------------
// Fragment load from an LDS tile with row stride 32 halves.
// 16-bit A/B fragment layout (ISA 7.12.2): lane holds row (lane&15); the 8
// dwords cover K pairs at kb = (vg/4)*16 + (lane/16)*8 + (vg%4)*2.
// ---------------------------------------------------------------------------
__device__ __forceinline__ v16h frag_ld32(const _Float16* rowp, int lane) {
    v16h f;
    unsigned int* fp = reinterpret_cast<unsigned int*>(&f);
    const int lh = (lane & 16) >> 1;  // 0 for lanes 0-15, 8 for lanes 16-31
#pragma unroll
    for (int vg = 0; vg < 8; ++vg) {
        const int kb = ((vg >> 2) << 4) + lh + ((vg & 3) << 1);
        fp[vg] = *reinterpret_cast<const unsigned int*>(rowp + kb);
    }
    return f;
}

// One K=32 step for a wave's 32x64 sub-tile: 2 A-frags x 4 B-frags -> 8 WMMAs.
__device__ __forceinline__ void mma_step(const _Float16* ldsA, const _Float16* ldsB,
                                         v8f c[2][4], int lane, int wm, int wn) {
    const int lrow = lane & 15;
    v16h a0 = frag_ld32(ldsA + (wm * 32 +      lrow) * 32, lane);
    v16h a1 = frag_ld32(ldsA + (wm * 32 + 16 + lrow) * 32, lane);
#pragma unroll
    for (int ni = 0; ni < 4; ++ni) {
        v16h b = frag_ld32(ldsB + (wn * 64 + ni * 16 + lrow) * 32, lane);
        c[0][ni] = __builtin_amdgcn_wmma_f32_16x16x32_f16(
            false, a0, false, b, (short)0, c[0][ni], false, false);
        c[1][ni] = __builtin_amdgcn_wmma_f32_16x16x32_f16(
            false, a1, false, b, (short)0, c[1][ni], false, false);
    }
}

// ---------------------------------------------------------------------------
// Shared GEMM core: C[128x128] += A[128xK] * B[NxK]^T  (both f16, [row][k]).
// Double-buffered LDS, one barrier per K-step, async prefetch of next tiles
// overlapped with WMMAs on current tiles.
// ---------------------------------------------------------------------------
__device__ __forceinline__ void gemm_core(v8f c[2][4],
                                          const _Float16* __restrict__ A, int lda,
                                          const _Float16* __restrict__ Bm, int ldb,
                                          int ksize, _Float16* ldsA, _Float16* ldsB,
                                          int t) {
    const int lane = t & 31, wid = t >> 5;
    const int wm = wid >> 1, wn = wid & 1;

#pragma unroll
    for (int i = 0; i < 2; ++i)
#pragma unroll
        for (int j = 0; j < 4; ++j)
#pragma unroll
            for (int r = 0; r < 8; ++r) c[i][j][r] = 0.0f;

    // Prime buffer 0.
    stage_tile(ldsA, A, lda, t);
    stage_tile(ldsB, Bm, ldb, t);

    const int nk = ksize >> 5;
    for (int ks = 0; ks < nk; ++ks) {
        async_wait_all();      // own DMA complete...
        __syncthreads();       // ...then everyone's: current buffer is ready
        const int cur = ks & 1;
        if (ks + 1 < nk) {     // prefetch next tiles into the other buffer
            stage_tile(ldsA + (cur ^ 1) * TILE_HALVES, A + ((ks + 1) << 5), lda, t);
            stage_tile(ldsB + (cur ^ 1) * TILE_HALVES, Bm + ((ks + 1) << 5), ldb, t);
        }
        mma_step(ldsA + cur * TILE_HALVES, ldsB + cur * TILE_HALVES, c, lane, wm, wn);
    }
}

// ---------------------------------------------------------------------------
// Kernel 0a: x (f32) -> xh (f16), elementwise.
// ---------------------------------------------------------------------------
__global__ __launch_bounds__(256) void cvt_x_kernel(const float* __restrict__ x,
                                                    _Float16* __restrict__ xh) {
    const size_t i = ((size_t)blockIdx.x * 256 + threadIdx.x) * 8;
#pragma unroll
    for (int j = 0; j < 8; ++j) xh[i + j] = (_Float16)x[i + j];
}

// ---------------------------------------------------------------------------
// Kernel 0b: W[k][n] (f32) -> WT[which][n][k] (f16), transpose + convert.
// ---------------------------------------------------------------------------
__global__ __launch_bounds__(256) void cvt_wT_kernel(const float* __restrict__ Wq,
                                                     const float* __restrict__ Wk,
                                                     const float* __restrict__ Wv,
                                                     _Float16* __restrict__ WT) {
    const int which = blockIdx.z;
    const float* W = (which == 0) ? Wq : (which == 1) ? Wk : Wv;
    const int idx = blockIdx.x * 256 + threadIdx.x;   // enumerates (n, k)
    const int n = idx >> 10, kk = idx & (DIM - 1);
    WT[(size_t)which * DIM * DIM + idx] = (_Float16)W[(size_t)kk * DIM + n];
}

// ---------------------------------------------------------------------------
// Kernel 1: QKV projection. out = xh[8192,1024] @ WT^T + bias.
// blockIdx.z: 0 -> q (scaled 1/sqrt(D)), 1 -> k, 2 -> v (stored transposed).
// ---------------------------------------------------------------------------
__global__ __launch_bounds__(256) void proj_kernel(
    const _Float16* __restrict__ xh, const _Float16* __restrict__ WT,
    const float* __restrict__ bq, const float* __restrict__ bk,
    const float* __restrict__ bv,
    _Float16* __restrict__ q, _Float16* __restrict__ k,
    _Float16* __restrict__ vT) {
    __shared__ alignas(16) _Float16 ldsA[2 * TILE_HALVES];
    __shared__ alignas(16) _Float16 ldsB[2 * TILE_HALVES];

    const int t = threadIdx.x, lane = t & 31, wid = t >> 5;
    const int wm = wid >> 1, wn = wid & 1;
    const int m0 = blockIdx.x * 128, n0 = blockIdx.y * 128;
    const int which = blockIdx.z;
    const float* bias = (which == 0) ? bq : (which == 1) ? bk : bv;

    v8f c[2][4];
    gemm_core(c, xh + (size_t)m0 * DIM, DIM,
              WT + (size_t)which * DIM * DIM + (size_t)n0 * DIM, DIM,
              DIM, ldsA, ldsB, t);

    const int mlo = (lane & 16) >> 1;
    const int nlo = lane & 15;
#pragma unroll
    for (int mi = 0; mi < 2; ++mi)
#pragma unroll
        for (int ni = 0; ni < 4; ++ni)
#pragma unroll
            for (int r = 0; r < 8; ++r) {
                const int m = m0 + wm * 32 + mi * 16 + mlo + r;
                const int n = n0 + wn * 64 + ni * 16 + nlo;
                float val = c[mi][ni][r] + bias[n];
                if (which == 0) {
                    q[(size_t)m * DIM + n] = (_Float16)(val * 0.03125f); // 1/sqrt(1024)
                } else if (which == 1) {
                    k[(size_t)m * DIM + n] = (_Float16)val;
                } else {
                    const int b = m >> 11, s = m & (SEQ - 1);
                    vT[((size_t)b * DIM + n) * SEQ + s] = (_Float16)val;
                }
            }
}

// ---------------------------------------------------------------------------
// Kernel 2: scores[b] = q[b] @ k[b]^T  (scale pre-folded into q).
// k stored [key][d] row-major is exactly the [N][K] layout B-frags want.
// ---------------------------------------------------------------------------
__global__ __launch_bounds__(256) void scores_kernel(
    const _Float16* __restrict__ q, const _Float16* __restrict__ k,
    float* __restrict__ scores) {
    __shared__ alignas(16) _Float16 ldsA[2 * TILE_HALVES];
    __shared__ alignas(16) _Float16 ldsB[2 * TILE_HALVES];

    const int t = threadIdx.x, lane = t & 31, wid = t >> 5;
    const int wm = wid >> 1, wn = wid & 1;
    const int m0 = blockIdx.x * 128, n0 = blockIdx.y * 128;
    const int b = blockIdx.z;

    float* sb = scores + (size_t)b * SEQ * SEQ;

    v8f c[2][4];
    gemm_core(c, q + (size_t)b * SEQ * DIM + (size_t)m0 * DIM, DIM,
              k + (size_t)b * SEQ * DIM + (size_t)n0 * DIM, DIM,
              DIM, ldsA, ldsB, t);

    const int mlo = (lane & 16) >> 1;
    const int nlo = lane & 15;
#pragma unroll
    for (int mi = 0; mi < 2; ++mi)
#pragma unroll
        for (int ni = 0; ni < 4; ++ni)
#pragma unroll
            for (int r = 0; r < 8; ++r) {
                const int m = m0 + wm * 32 + mi * 16 + mlo + r;
                const int n = n0 + wn * 64 + ni * 16 + nlo;
                sb[(size_t)m * SEQ + n] = c[mi][ni][r];
            }
}

// ---------------------------------------------------------------------------
// Kernel 3: row softmax. One wave per row (2048 f32 = 64 regs/lane).
// Writes f16 probs into the FRONT HALF of the same f32 row (rows disjoint;
// within a wave all reads precede all writes -> race free).
// ---------------------------------------------------------------------------
__global__ __launch_bounds__(256) void softmax_kernel(float* __restrict__ scores) {
    const int gw = (int)((blockIdx.x * blockDim.x + threadIdx.x) >> 5); // row id
    const int lane = threadIdx.x & 31;
    float* row = scores + (size_t)gw * SEQ;

    float vals[64];
    float mx = -1e30f;
#pragma unroll
    for (int j = 0; j < 64; ++j) {
        vals[j] = row[lane + 32 * j];
        mx = fmaxf(mx, vals[j]);
    }
#pragma unroll
    for (int off = 16; off >= 1; off >>= 1)
        mx = fmaxf(mx, __shfl_xor(mx, off, 32));

    float sum = 0.0f;
#pragma unroll
    for (int j = 0; j < 64; ++j) {
        vals[j] = __expf(vals[j] - mx);
        sum += vals[j];
    }
#pragma unroll
    for (int off = 16; off >= 1; off >>= 1)
        sum += __shfl_xor(sum, off, 32);

    const float inv = 1.0f / sum;
    _Float16* prow = reinterpret_cast<_Float16*>(row);
#pragma unroll
    for (int j = 0; j < 64; ++j)
        prow[lane + 32 * j] = (_Float16)(vals[j] * inv);
}

// ---------------------------------------------------------------------------
// Kernel 4: y[b] = P[b] @ V[b].  P rows are f16 with stride 2*SEQ halves
// (front half of each f32 score row). vT is [d][key] so B-frags are direct.
// ---------------------------------------------------------------------------
__global__ __launch_bounds__(256) void out_kernel(
    const float* __restrict__ scores, const _Float16* __restrict__ vT,
    float* __restrict__ y) {
    __shared__ alignas(16) _Float16 ldsA[2 * TILE_HALVES];
    __shared__ alignas(16) _Float16 ldsB[2 * TILE_HALVES];

    const int t = threadIdx.x, lane = t & 31, wid = t >> 5;
    const int wm = wid >> 1, wn = wid & 1;
    const int m0 = blockIdx.x * 128;   // query rows
    const int n0 = blockIdx.y * 128;   // output dim cols
    const int b = blockIdx.z;

    const _Float16* Pb = reinterpret_cast<const _Float16*>(scores + (size_t)b * SEQ * SEQ);
    const int ldP = 2 * SEQ;           // halves per P row
    float* yb = y + (size_t)b * SEQ * DIM;

    v8f c[2][4];
    gemm_core(c, Pb + (size_t)m0 * ldP, ldP,
              vT + (size_t)b * DIM * SEQ + (size_t)n0 * SEQ, SEQ,
              SEQ, ldsA, ldsB, t);

    const int mlo = (lane & 16) >> 1;
    const int nlo = lane & 15;
#pragma unroll
    for (int mi = 0; mi < 2; ++mi)
#pragma unroll
        for (int ni = 0; ni < 4; ++ni)
#pragma unroll
            for (int r = 0; r < 8; ++r) {
                const int m = m0 + wm * 32 + mi * 16 + mlo + r;
                const int n = n0 + wn * 64 + ni * 16 + nlo;
                yb[(size_t)m * DIM + n] = c[mi][ni][r];
            }
}

// ---------------------------------------------------------------------------
// Launch.  Workspace layout (bytes):
//   [0,16M)      q   f16 [B][S][D]
//   [16M,32M)    k   f16 [B][S][D]
//   [32M,48M)    vT  f16 [B][D][S]
//   [48M,112M)   scores f32 [B][S][S] (softmax rewrites each row's front half
//                                      as f16 probabilities)
//   [112M,128M)  xh  f16 [B*S][D]
//   [128M,134M)  WT  f16 [3][D][D]   (q,k,v weights transposed)
// ---------------------------------------------------------------------------
extern "C" void kernel_launch(void* const* d_in, const int* in_sizes, int n_in,
                              void* d_out, int out_size, void* d_ws, size_t ws_size,
                              hipStream_t stream) {
    const float* x  = (const float*)d_in[0];
    const float* Wk = (const float*)d_in[1];
    const float* bk = (const float*)d_in[2];
    const float* Wq = (const float*)d_in[3];
    const float* bq = (const float*)d_in[4];
    const float* Wv = (const float*)d_in[5];
    const float* bv = (const float*)d_in[6];

    char* ws = (char*)d_ws;
    const size_t qkvBytes = (size_t)BATCH * SEQ * DIM * sizeof(_Float16); // 16 MB
    _Float16* q   = (_Float16*)(ws);
    _Float16* k   = (_Float16*)(ws + qkvBytes);
    _Float16* vT  = (_Float16*)(ws + 2 * qkvBytes);
    float* scores = (float*)(ws + 3 * qkvBytes);
    const size_t scBytes = (size_t)BATCH * SEQ * SEQ * sizeof(float);    // 64 MB
    _Float16* xh  = (_Float16*)(ws + 3 * qkvBytes + scBytes);
    _Float16* WT  = (_Float16*)(ws + 3 * qkvBytes + scBytes + qkvBytes);

    // 0) precision conversion / weight transpose (bandwidth-trivial)
    cvt_x_kernel<<<dim3((BATCH * SEQ * DIM) / (256 * 8)), 256, 0, stream>>>(x, xh);
    cvt_wT_kernel<<<dim3((DIM * DIM) / 256, 1, 3), 256, 0, stream>>>(Wq, Wk, Wv, WT);

    // 1) q/k/v projections (grid.z picks which weight / output layout)
    proj_kernel<<<dim3(BATCH * SEQ / 128, DIM / 128, 3), 256, 0, stream>>>(
        xh, WT, bq, bk, bv, q, k, vT);

    // 2) scores = q @ k^T (pre-scaled)
    scores_kernel<<<dim3(SEQ / 128, SEQ / 128, BATCH), 256, 0, stream>>>(q, k, scores);

    // 3) row softmax, f32 -> f16 in place
    softmax_kernel<<<dim3(BATCH * SEQ / 8), 256, 0, stream>>>(scores);

    // 4) y = P @ V
    out_kernel<<<dim3(SEQ / 128, DIM / 128, BATCH), 256, 0, stream>>>(
        scores, vT, (float*)d_out);
}
